// GNNSCFDN_32289564131434
// MI455X (gfx1250) — compile-verified, matrix operand order checked by reading, not run on hardware
//
#include <hip/hip_runtime.h>
#include <hip/hip_bf16.h>
#include <math.h>

#define N_NODES 1024
#define NB      32
#define NE      8192
#define DIMO    64
#define IN_DIM  66
#define COLS    4224   // DIMO * IN_DIM

typedef __attribute__((ext_vector_type(16))) __bf16 v16bf;
typedef __attribute__((ext_vector_type(8)))  float  v8f;
typedef int v4i __attribute__((vector_size(16)));
typedef __attribute__((address_space(1))) v4i* gv4i_p;   // global int4*
typedef __attribute__((address_space(3))) v4i* lv4i_p;   // LDS int4*

__device__ __forceinline__ float sigmoidf_(float x) { return 1.f / (1.f + expf(-x)); }
__device__ __forceinline__ float siluf_(float x)    { return x * sigmoidf_(x); }

// ---------------- Stage A: EGNN edge MLP (e1->e2->c1->c2), segment sums via atomics
__global__ void k_edge(const float* mol, const int* ei,
                       const float* e1w, const float* e1b,
                       const float* e2w, const float* e2b,
                       const float* c1w, const float* c1b,
                       const float* c2w, const float* c2b,
                       float* coors_acc, float* mi_acc, float* deg) {
  __shared__ float in127[128];
  __shared__ float rel[3];
  __shared__ float eh[254];
  __shared__ float mij[63];
  __shared__ float ch[252];
  int e = blockIdx.x, tid = threadIdx.x;
  int src = ei[e], dst = ei[NE + e];
  if (tid < 63) {
    in127[tid]      = mol[dst * IN_DIM + 3 + tid];
    in127[63 + tid] = mol[src * IN_DIM + 3 + tid];
  }
  if (tid == 0) {
    float r2 = 0.f;
    for (int d = 0; d < 3; ++d) {
      float r = mol[src * IN_DIM + d] - mol[dst * IN_DIM + d];
      rel[d] = r; r2 += r * r;
    }
    in127[126] = r2;
  }
  __syncthreads();
  if (tid < 254) {
    float acc = e1b[tid];
    const float* w = e1w + tid * 127;
    for (int i = 0; i < 127; ++i) acc += w[i] * in127[i];
    eh[tid] = siluf_(acc);
  }
  __syncthreads();
  if (tid < 63) {
    float acc = e2b[tid];
    const float* w = e2w + tid * 254;
    for (int i = 0; i < 254; ++i) acc += w[i] * eh[i];
    float v = siluf_(acc);
    mij[tid] = v;
    atomicAdd(&mi_acc[dst * 63 + tid], v);
  }
  __syncthreads();
  if (tid < 252) {
    float acc = c1b[tid];
    const float* w = c1w + tid * 63;
    for (int i = 0; i < 63; ++i) acc += w[i] * mij[i];
    ch[tid] = siluf_(acc);
  }
  __syncthreads();
  if (tid == 0) {
    float cw = c2b[0];
    for (int i = 0; i < 252; ++i) cw += c2w[i] * ch[i];
    for (int d = 0; d < 3; ++d) atomicAdd(&coors_acc[dst * 3 + d], cw * rel[d]);
    atomicAdd(&deg[dst], 1.0f);
  }
}

// ---------------- Stage A: node MLP (n1->n2), build smi = cat(coors_out, hid), NaN->0
__global__ void k_node(const float* mol, const float* mi_acc, const float* coors_acc,
                       const float* n1w, const float* n1b,
                       const float* n2w, const float* n2b, float* smi) {
  __shared__ float in126[126];
  __shared__ float t126[126];
  int n = blockIdx.x, tid = threadIdx.x;   // 128 threads
  if (tid < 63) {
    in126[tid]      = mol[n * IN_DIM + 3 + tid];
    in126[63 + tid] = mi_acc[n * 63 + tid];
  }
  __syncthreads();
  if (tid < 126) {
    float acc = n1b[tid];
    const float* w = n1w + tid * 126;
    for (int i = 0; i < 126; ++i) acc += w[i] * in126[i];
    t126[tid] = siluf_(acc);
  }
  __syncthreads();
  if (tid < 63) {
    float acc = n2b[tid];
    const float* w = n2w + tid * 126;
    for (int i = 0; i < 126; ++i) acc += w[i] * t126[i];
    float hid = in126[tid] + acc;
    if (hid != hid) hid = 0.f;
    smi[n * IN_DIM + 3 + tid] = hid;
  }
  if (tid < 3) {
    float v = mol[n * IN_DIM + tid] + coors_acc[n * 3 + tid];
    if (v != v) v = 0.f;
    smi[n * IN_DIM + tid] = v;
  }
}

// ---------------- BatchNorm over axis 0 (population stats)
__global__ void k_colstat(const float* smi, float* stat) {
  __shared__ float s1[256], s2[256];
  int c = blockIdx.x, tid = threadIdx.x;
  float a = 0.f, b = 0.f;
  for (int r = tid; r < N_NODES; r += 256) {
    float v = smi[r * IN_DIM + c];
    a += v; b += v * v;
  }
  s1[tid] = a; s2[tid] = b;
  __syncthreads();
  for (int s = 128; s > 0; s >>= 1) {
    if (tid < s) { s1[tid] += s1[tid + s]; s2[tid] += s2[tid + s]; }
    __syncthreads();
  }
  if (tid == 0) {
    float m = s1[0] / (float)N_NODES;
    stat[c * 2] = m;
    stat[c * 2 + 1] = s2[0] / (float)N_NODES - m * m;
  }
}

__global__ void k_bn(const float* smi, const float* stat, const float* g,
                     const float* bb, float* h) {
  int idx = blockIdx.x * 256 + threadIdx.x;
  if (idx >= N_NODES * IN_DIM) return;
  int c = idx % IN_DIM;
  float m = stat[c * 2], v = stat[c * 2 + 1];
  h[idx] = g[c] * (smi[idx] - m) * rsqrtf(v + 1e-5f) + bb[c];
}

// ---------------- Transposed adjacency in bf16 (0/1 exact): A[a][b] = adj[b][a]
__global__ void k_adjT(const float* adj, __bf16* adjT) {
  int idx = blockIdx.x * 256 + threadIdx.x;
  if (idx >= N_NODES * N_NODES) return;
  int a = idx / N_NODES, b = idx % N_NODES;
  adjT[idx] = (__bf16)adj[b * N_NODES + a];
}

// ---------------- Loop: message passing, W_e recomputed on the fly (nn_w is L2-resident)
__global__ void k_msg(const float* h, const int* ei, const float* ea,
                      const float* nnw, const float* nnb, float* agg) {
  __shared__ float hv[IN_DIM];
  __shared__ float av[10];
  int e = blockIdx.x, tid = threadIdx.x;   // 96 threads
  int src = ei[e], dst = ei[NE + e];
  if (tid < IN_DIM) hv[tid] = h[src * IN_DIM + tid];
  if (tid < 10)     av[tid] = ea[e * 10 + tid];
  __syncthreads();
  if (tid < IN_DIM) {
    float acc = 0.f;
    for (int i = 0; i < IN_DIM; ++i) {
      int r = i * IN_DIM + tid;
      const float* w = nnw + r * 10;
      float Wv = nnb[r];
      for (int f = 0; f < 10; ++f) Wv += w[f] * av[f];
      acc += hv[i] * Wv;
    }
    atomicAdd(&agg[dst * IN_DIM + tid], acc);
  }
}

__global__ void k_mnode(const float* h, const float* agg, const float* deg,
                        const float* root, const float* nn_bias, float* mvec) {
  __shared__ float hv[IN_DIM];
  int n = blockIdx.x, tid = threadIdx.x;   // 96 threads
  if (tid < IN_DIM) hv[tid] = h[n * IN_DIM + tid];
  __syncthreads();
  if (tid < IN_DIM) {
    float d = deg[n]; if (d < 1.f) d = 1.f;
    float acc = agg[n * IN_DIM + tid] / d + nn_bias[tid];
    for (int i = 0; i < IN_DIM; ++i) acc += hv[i] * root[i * IN_DIM + tid];
    mvec[n * IN_DIM + tid] = acc > 0.f ? acc : 0.f;
  }
}

// ---------------- Build t0 TRANSPOSED: t0T[col][b]  (K-contiguous per GEMM column)
__global__ void k_t0(const float* mvec, const float* conv1w, int it, __bf16* t0T) {
  int idx = blockIdx.x * 256 + threadIdx.x;
  if (idx >= N_NODES * COLS) return;
  int b = idx & (N_NODES - 1);
  int col = idx >> 10;
  int o = col / IN_DIM, c = col % IN_DIM;
  const float* w = conv1w + ((it * DIMO + o) * N_NODES + b) * 3;
  float acc = 0.f;
  for (int k = 0; k < 3; ++k) {
    int cc = c + k - 1;
    if (cc >= 0 && cc < IN_DIM) acc += w[k] * mvec[b * IN_DIM + cc];
  }
  t0T[idx] = (__bf16)acc;
}

// ---------------- The big GEMM on WMMA: mid[a][col] = sum_b adjT[a][b]*t0T[col][b] + conv1_b
// Both operands are K-contiguous, so each lane's ISA fragment = two 16B runs:
//   element j (0..15) -> K = (j>>3)*16 + (lane>=16)*8 + (j&7)
// i.e. uint4 chunks {k8+half, k8+2+half}. Four global_load_b128 feed one v_wmma.
__global__ void __launch_bounds__(256) k_gemm(const __bf16* __restrict__ adjT,
                                              const __bf16* __restrict__ t0T,
                                              const float* conv1b, int it,
                                              float* __restrict__ mid) {
  int tid = threadIdx.x;
  int colBase = blockIdx.x * 128;
  int rowBase = blockIdx.y * 16;
  int lane = tid & 31, wave = tid >> 5;
  int half = lane >> 4, l15 = lane & 15;
  const uint4* Ap = (const uint4*)(adjT + (size_t)(rowBase + l15) * N_NODES);
  const uint4* Bp = (const uint4*)(t0T + (size_t)(colBase + wave * 16 + l15) * N_NODES);
  union Frag { uint4 q[2]; v16bf v; };
  v8f acc = {};
#pragma unroll 4
  for (int k8 = 0; k8 < N_NODES / 8; k8 += 4) {   // one WMMA consumes K=32 = 4 uint4
    if (k8 + 4 < N_NODES / 8)                     // global_prefetch_b8 of next B chunk
      __builtin_prefetch(Bp + k8 + 4, 0, 1);
    Frag fa, fb;
    fa.q[0] = Ap[k8 + half];
    fa.q[1] = Ap[k8 + 2 + half];
    fb.q[0] = Bp[k8 + half];
    fb.q[1] = Bp[k8 + 2 + half];
    acc = __builtin_amdgcn_wmma_f32_16x16x32_bf16(false, fa.v, false, fb.v,
                                                  (short)0, acc, false, false);
  }
#pragma unroll
  for (int r = 0; r < 8; ++r) {
    int m = r + (half << 3);                      // ISA f32 C/D layout
    int col = colBase + wave * 16 + l15;
    mid[(size_t)(rowBase + m) * COLS + col] = acc[r] + conv1b[it * DIMO + col / IN_DIM];
  }
}

// ---------------- conv2 (1 out-channel, 3-tap over o); row staged via async global->LDS
__global__ void k_sp(const float* mid, const float* conv2w, const float* conv2b,
                     int it, float* sp) {
  __shared__ __align__(16) float mrow[COLS];
  int n = blockIdx.x, tid = threadIdx.x;   // 64 threads
#if __has_builtin(__builtin_amdgcn_global_load_async_to_lds_b128) && \
    __has_builtin(__builtin_amdgcn_s_wait_asynccnt)
  {
    const uint4* gsrc = (const uint4*)(mid + (size_t)n * COLS);
    uint4* ldst = (uint4*)mrow;
    for (int j = tid; j < COLS / 4; j += 64)
      __builtin_amdgcn_global_load_async_to_lds_b128(
          (gv4i_p)(gsrc + j), (lv4i_p)(ldst + j), 0, 0);
    __builtin_amdgcn_s_wait_asynccnt(0);
  }
  __syncthreads();
#else
  for (int j = tid; j < COLS; j += 64) mrow[j] = mid[(size_t)n * COLS + j];
  __syncthreads();
#endif
  float acc = conv2b[it];
  const float* w = conv2w + it * IN_DIM * 3;
  for (int c = 0; c < IN_DIM; ++c)
    for (int k = 0; k < 3; ++k) {
      int oo = tid + k - 1;
      if (oo >= 0 && oo < DIMO) acc += mrow[oo * IN_DIM + c] * w[c * 3 + k];
    }
  sp[n * DIMO + tid] = acc;
}

// ---------------- cm = cat(m, sp); f = leaky_relu(linf(cm))
__global__ void k_cmf(const float* mvec, const float* sp, const float* lfw,
                      const float* lfb, float* cm, float* fvec) {
  __shared__ float cmv[130];
  __shared__ float red[128];
  int n = blockIdx.x, tid = threadIdx.x;   // 128 threads
  if (tid < IN_DIM) cmv[tid] = mvec[n * IN_DIM + tid];
  if (tid < DIMO)   cmv[IN_DIM + tid] = sp[n * DIMO + tid];
  __syncthreads();
  float p = 0.f;
  for (int j = tid; j < 130; j += 128) p += lfw[j] * cmv[j];
  red[tid] = p;
  __syncthreads();
  for (int s = 64; s > 0; s >>= 1) {
    if (tid < s) red[tid] += red[tid + s];
    __syncthreads();
  }
  if (tid == 0) {
    float f = red[0] + lfb[0];
    fvec[n] = f >= 0.f ? f : 0.01f * f;
  }
  for (int j = tid; j < 130; j += 128) cm[n * 130 + j] = cmv[j];
}

// ---------------- segment softmax (contiguous groups of 32 nodes per batch)
__global__ void k_soft(const float* ein, float* aout) {
  __shared__ float v[32], w[32];
  int b = blockIdx.x, t = threadIdx.x;     // 32 threads (one wave)
  int n = b * 32 + t;
  v[t] = ein[n];
  __syncthreads();
  float mx = v[0];
  for (int i = 1; i < 32; ++i) mx = fmaxf(mx, v[i]);
  float ex = expf(v[t] - mx);
  w[t] = ex;
  __syncthreads();
  float s = 0.f;
  for (int i = 0; i < 32; ++i) s += w[i];
  aout[n] = ex / (s + 1e-16f);
}

// ---------------- GRU node update (x = af*cm)
__global__ void k_gru(const float* cm, const float* af, float* h,
                      const float* wih, const float* whh,
                      const float* bih, const float* bhh) {
  __shared__ float xv[130], hv[IN_DIM], gi[198], gh[198];
  int n = blockIdx.x, tid = threadIdx.x;   // 256 threads
  float a = af[n];
  if (tid < 130)    xv[tid] = a * cm[n * 130 + tid];
  if (tid < IN_DIM) hv[tid] = h[n * IN_DIM + tid];
  __syncthreads();
  if (tid < 198) {
    float acc = bih[tid];
    const float* w = wih + tid * 130;
    for (int j = 0; j < 130; ++j) acc += w[j] * xv[j];
    gi[tid] = acc;
    acc = bhh[tid];
    const float* w2 = whh + tid * IN_DIM;
    for (int j = 0; j < IN_DIM; ++j) acc += w2[j] * hv[j];
    gh[tid] = acc;
  }
  __syncthreads();
  if (tid < IN_DIM) {
    float r = sigmoidf_(gi[tid] + gh[tid]);
    float z = sigmoidf_(gi[IN_DIM + tid] + gh[IN_DIM + tid]);
    float nn = tanhf(gi[2 * IN_DIM + tid] + r * gh[2 * IN_DIM + tid]);
    h[n * IN_DIM + tid] = (1.f - z) * nn + z * hv[tid];
  }
}

// ---------------- Set2Set LSTM cell
__global__ void k_lstm(const float* qstar, float* hs, float* cs,
                       const float* wih, const float* whh,
                       const float* bih, const float* bhh) {
  __shared__ float qv[132], hv[IN_DIM], g[264];
  int b = blockIdx.x, tid = threadIdx.x;   // 256 threads
  if (tid < 132)    qv[tid] = qstar[b * 132 + tid];
  if (tid < IN_DIM) hv[tid] = hs[b * IN_DIM + tid];
  __syncthreads();
  for (int o = tid; o < 264; o += 256) {
    float acc = bih[o] + bhh[o];
    const float* w = wih + o * 132;
    for (int j = 0; j < 132; ++j) acc += w[j] * qv[j];
    const float* w2 = whh + o * IN_DIM;
    for (int j = 0; j < IN_DIM; ++j) acc += w2[j] * hv[j];
    g[o] = acc;
  }
  __syncthreads();
  if (tid < IN_DIM) {
    float ig = sigmoidf_(g[tid]);
    float fg = sigmoidf_(g[IN_DIM + tid]);
    float gg = tanhf(g[2 * IN_DIM + tid]);
    float og = sigmoidf_(g[3 * IN_DIM + tid]);
    float c = fg * cs[b * IN_DIM + tid] + ig * gg;
    cs[b * IN_DIM + tid] = c;
    hs[b * IN_DIM + tid] = og * tanhf(c);
  }
}

__global__ void k_e(const float* h, const float* hs, const int* mb, float* evec) {
  int n = blockIdx.x * blockDim.x + threadIdx.x;
  if (n >= N_NODES) return;
  int b = mb[n];
  float acc = 0.f;
  for (int d = 0; d < IN_DIM; ++d) acc += h[n * IN_DIM + d] * hs[b * IN_DIM + d];
  evec[n] = acc;
}

__global__ void k_rs(const float* aa, const float* h, const float* hs, float* qstar) {
  int b = blockIdx.x, tid = threadIdx.x;   // 96 threads
  if (tid < IN_DIM) {
    float acc = 0.f;
    for (int i = 0; i < 32; ++i) {
      int n = b * 32 + i;
      acc += aa[n] * h[n * IN_DIM + tid];
    }
    qstar[b * 132 + tid]          = hs[b * IN_DIM + tid];
    qstar[b * 132 + IN_DIM + tid] = acc;
  }
}

// ---------------- fingers MLP
__global__ void k_fw1(const float* fing, const float* w, const float* bias, float* o1) {
  __shared__ float fin[1024];
  int b = blockIdx.x, tid = threadIdx.x;
  for (int j = tid; j < 1024; j += 256) fin[j] = fing[b * 1024 + j];
  __syncthreads();
  for (int o = tid; o < 512; o += 256) {
    float acc = bias[o];
    const float* wr = w + o * 1024;
    for (int j = 0; j < 1024; ++j) acc += wr[j] * fin[j];
    o1[b * 512 + o] = acc > 0.f ? acc : 0.f;
  }
}
__global__ void k_fw2(const float* in1, const float* w, const float* bias, float* o2) {
  __shared__ float fin[512];
  int b = blockIdx.x, tid = threadIdx.x;
  for (int j = tid; j < 512; j += 256) fin[j] = in1[b * 512 + j];
  __syncthreads();
  if (tid < 256) {
    float acc = bias[tid];
    const float* wr = w + tid * 512;
    for (int j = 0; j < 512; ++j) acc += wr[j] * fin[j];
    o2[b * 256 + tid] = acc > 0.f ? acc : 0.f;
  }
}
__global__ void k_fw3(const float* in2, const float* w, const float* bias, float* o3) {
  __shared__ float fin[256];
  int b = blockIdx.x, tid = threadIdx.x;   // 64 threads
  for (int j = tid; j < 256; j += 64) fin[j] = in2[b * 256 + j];
  __syncthreads();
  float acc = bias[tid];
  const float* wr = w + tid * 256;
  for (int j = 0; j < 256; ++j) acc += wr[j] * fin[j];
  o3[b * DIMO + tid] = acc;
}

__global__ void k_final(const float* qstar, const float* fw3, const float* pw,
                        const float* pb, float* out) {
  int b = threadIdx.x;   // 32 threads
  float acc = pb[0];
  for (int j = 0; j < 132; ++j) acc += pw[j] * qstar[b * 132 + j];
  for (int j = 0; j < DIMO; ++j) acc += pw[132 + j] * fw3[b * DIMO + j];
  out[b] = acc;
}

extern "C" void kernel_launch(void* const* d_in, const int* in_sizes, int n_in,
                              void* d_out, int out_size, void* d_ws, size_t ws_size,
                              hipStream_t stream) {
  (void)in_sizes; (void)n_in; (void)out_size; (void)ws_size;
  const float* mol  = (const float*)d_in[0];
  const float* ea   = (const float*)d_in[1];
  const float* fing = (const float*)d_in[2];
  const int*   eidx = (const int*)d_in[3];
  const int*   mb   = (const int*)d_in[4];
  const float* adj  = (const float*)d_in[5];
  const float *e1w = (const float*)d_in[6],  *e1b = (const float*)d_in[7];
  const float *e2w = (const float*)d_in[8],  *e2b = (const float*)d_in[9];
  const float *c1w = (const float*)d_in[10], *c1b = (const float*)d_in[11];
  const float *c2w = (const float*)d_in[12], *c2b = (const float*)d_in[13];
  const float *n1w = (const float*)d_in[14], *n1b = (const float*)d_in[15];
  const float *n2w = (const float*)d_in[16], *n2b = (const float*)d_in[17];
  const float *bng = (const float*)d_in[18], *bnb = (const float*)d_in[19];
  const float *nnw = (const float*)d_in[20], *nnb = (const float*)d_in[21];
  const float *root = (const float*)d_in[22], *nnbias = (const float*)d_in[23];
  const float *cv1w = (const float*)d_in[24], *cv1b = (const float*)d_in[25];
  const float *cv2w = (const float*)d_in[26], *cv2b = (const float*)d_in[27];
  const float *lfw = (const float*)d_in[28], *lfb = (const float*)d_in[29];
  const float *gwih = (const float*)d_in[30], *gwhh = (const float*)d_in[31];
  const float *gbih = (const float*)d_in[32], *gbhh = (const float*)d_in[33];
  const float *lwih = (const float*)d_in[34], *lwhh = (const float*)d_in[35];
  const float *lbih = (const float*)d_in[36], *lbhh = (const float*)d_in[37];
  const float *d1w = (const float*)d_in[38], *d1b = (const float*)d_in[39];
  const float *d2w = (const float*)d_in[40], *d2b = (const float*)d_in[41];
  const float *d3w = (const float*)d_in[42], *d3b = (const float*)d_in[43];
  const float *pw = (const float*)d_in[44], *pb = (const float*)d_in[45];
  float* outp = (float*)d_out;

  char* base = (char*)d_ws;
  size_t off = 0;
  auto alloc = [&](size_t bytes) -> void* {
    void* r = base + off;
    off = (off + bytes + 255) & ~(size_t)255;
    return r;
  };
  __bf16* adjT  = (__bf16*)alloc((size_t)N_NODES * N_NODES * 2);
  __bf16* t0T   = (__bf16*)alloc((size_t)N_NODES * COLS * 2);
  float*  mid   = (float*)alloc((size_t)N_NODES * COLS * 4);
  float*  cacc  = (float*)alloc(N_NODES * 3 * 4);
  float*  miacc = (float*)alloc(N_NODES * 63 * 4);
  float*  deg   = (float*)alloc(N_NODES * 4);
  float*  smi   = (float*)alloc(N_NODES * IN_DIM * 4);
  float*  stat  = (float*)alloc(IN_DIM * 2 * 4);
  float*  h     = (float*)alloc(N_NODES * IN_DIM * 4);
  float*  agg   = (float*)alloc(N_NODES * IN_DIM * 4);
  float*  mvec  = (float*)alloc(N_NODES * IN_DIM * 4);
  float*  sp    = (float*)alloc(N_NODES * DIMO * 4);
  float*  cm    = (float*)alloc(N_NODES * 130 * 4);
  float*  fvec  = (float*)alloc(N_NODES * 4);
  float*  af    = (float*)alloc(N_NODES * 4);
  float*  hs    = (float*)alloc(NB * IN_DIM * 4);
  float*  cs    = (float*)alloc(NB * IN_DIM * 4);
  float*  qstar = (float*)alloc(NB * 132 * 4);
  float*  evec  = (float*)alloc(N_NODES * 4);
  float*  aa    = (float*)alloc(N_NODES * 4);
  float*  fo1   = (float*)alloc(NB * 512 * 4);
  float*  fo2   = (float*)alloc(NB * 256 * 4);
  float*  fo3   = (float*)alloc(NB * DIMO * 4);

  (void)hipMemsetAsync(cacc,  0, N_NODES * 3 * 4, stream);
  (void)hipMemsetAsync(miacc, 0, N_NODES * 63 * 4, stream);
  (void)hipMemsetAsync(deg,   0, N_NODES * 4, stream);
  (void)hipMemsetAsync(hs,    0, NB * IN_DIM * 4, stream);
  (void)hipMemsetAsync(cs,    0, NB * IN_DIM * 4, stream);
  (void)hipMemsetAsync(qstar, 0, NB * 132 * 4, stream);

  k_edge<<<NE, 256, 0, stream>>>(mol, eidx, e1w, e1b, e2w, e2b, c1w, c1b, c2w, c2b,
                                 cacc, miacc, deg);
  k_node<<<N_NODES, 128, 0, stream>>>(mol, miacc, cacc, n1w, n1b, n2w, n2b, smi);
  k_colstat<<<IN_DIM, 256, 0, stream>>>(smi, stat);
  k_bn<<<(N_NODES * IN_DIM + 255) / 256, 256, 0, stream>>>(smi, stat, bng, bnb, h);
  k_adjT<<<(N_NODES * N_NODES + 255) / 256, 256, 0, stream>>>(adj, adjT);

  for (int it = 0; it < 3; ++it) {
    (void)hipMemsetAsync(agg, 0, N_NODES * IN_DIM * 4, stream);
    k_msg<<<NE, 96, 0, stream>>>(h, eidx, ea, nnw, nnb, agg);
    k_mnode<<<N_NODES, 96, 0, stream>>>(h, agg, deg, root, nnbias, mvec);
    k_t0<<<(N_NODES * COLS + 255) / 256, 256, 0, stream>>>(mvec, cv1w, it, t0T);
    k_gemm<<<dim3(COLS / 128, N_NODES / 16), 256, 0, stream>>>(adjT, t0T, cv1b, it, mid);
    k_sp<<<N_NODES, 64, 0, stream>>>(mid, cv2w, cv2b, it, sp);
    k_cmf<<<N_NODES, 128, 0, stream>>>(mvec, sp, lfw, lfb, cm, fvec);
    k_soft<<<NB, 32, 0, stream>>>(fvec, af);
    k_gru<<<N_NODES, 256, 0, stream>>>(cm, af, h, gwih, gwhh, gbih, gbhh);
  }

  for (int t = 0; t < 3; ++t) {
    k_lstm<<<NB, 256, 0, stream>>>(qstar, hs, cs, lwih, lwhh, lbih, lbhh);
    k_e<<<(N_NODES + 255) / 256, 256, 0, stream>>>(h, hs, mb, evec);
    k_soft<<<NB, 32, 0, stream>>>(evec, aa);
    k_rs<<<NB, 96, 0, stream>>>(aa, h, hs, qstar);
  }

  k_fw1<<<NB, 256, 0, stream>>>(fing, d1w, d1b, fo1);
  k_fw2<<<NB, 256, 0, stream>>>(fo1, d2w, d2b, fo2);
  k_fw3<<<NB, 64, 0, stream>>>(fo2, d3w, d3b, fo3);
  k_final<<<1, 32, 0, stream>>>(qstar, fo3, pw, pb, outp);
}
